// MultiHeadAttention_88012469829954
// MI455X (gfx1250) — matrix operand order repaired and checked by reading, not compiled, not run on hardware
//
#include <hip/hip_runtime.h>

typedef __attribute__((ext_vector_type(16))) __bf16 v16bf;
typedef __attribute__((ext_vector_type(8)))  float  v8f;
typedef __attribute__((ext_vector_type(8)))  unsigned short ushort8;
typedef __attribute__((__vector_size__(16))) int i32x4;

#define DEV static __device__ __forceinline__

// ---------- helpers ----------
DEV unsigned short f2bf(float f) {
    unsigned int u = __float_as_uint(f);
    u += 0x7FFFu + ((u >> 16) & 1u);       // round-to-nearest-even
    return (unsigned short)(u >> 16);
}

union FragBF {
    v16bf v;
    ushort8 h[2];
    unsigned short u[16];
};

// A-matrix 16x32 bf16 fragment (ISA 7.12.2): lane -> row = lane&15,
// kb = lane<16 ? 0 : 8; VGPR0-3 hold K=kb..kb+7, VGPR4-7 hold K=16+kb..16+kb+7.
DEV v16bf load_a_bf16(const unsigned short* A, int lda, int lane) {
    int row = lane & 15;
    int kb  = (lane < 16) ? 0 : 8;
    FragBF f;
    f.h[0] = *(const ushort8*)(A + (size_t)row * lda + kb);
    f.h[1] = *(const ushort8*)(A + (size_t)row * lda + 16 + kb);
    return f.v;
}

// B-matrix 32x16 bf16 fragment: lane -> col = lane&15, kb = lane<16 ? 0 : 8;
// holds K=kb..kb+7, then K=16+kb..16+kb+7. Source is column-contiguous: Bt[N][K] row-major.
DEV v16bf load_b_bf16(const unsigned short* Bt, int ldb, int lane) {
    int col = lane & 15;
    int kb  = (lane < 16) ? 0 : 8;
    FragBF f;
    f.h[0] = *(const ushort8*)(Bt + (size_t)col * ldb + kb);
    f.h[1] = *(const ushort8*)(Bt + (size_t)col * ldb + kb + 16);
    return f.v;
}

DEV v8f wmma_bf16(v16bf a, v16bf b, v8f c) {
    return __builtin_amdgcn_wmma_f32_16x16x32_bf16(
        false, a, false, b, (short)0, c, false, false);
}

// ---------- async global -> LDS copy (16 bytes) ----------
// Builtin signature (from hipcc diagnostic): param1 = i32x4 in AS1 (global),
// param2 = i32x4 in AS3 (LDS), then two immediates (offset, cpol).
// Generic LDS addresses truncate to the 32-bit DS offset (aperture rule);
// generic global addresses are identity in AS1, so integer casts are safe.
DEV void async_copy16(unsigned short* ldst, const unsigned short* gsrc) {
#if defined(__has_builtin) && __has_builtin(__builtin_amdgcn_global_load_async_to_lds_b128)
    __builtin_amdgcn_global_load_async_to_lds_b128(
        (__attribute__((address_space(1))) i32x4*)(unsigned long long)gsrc,
        (__attribute__((address_space(3))) i32x4*)(unsigned int)(unsigned long long)ldst,
        0, 0);
#else
    *(ushort8*)ldst = *(const ushort8*)gsrc;
#endif
}

DEV void wait_async0() {
#if defined(__has_builtin) && __has_builtin(__builtin_amdgcn_s_wait_asynccnt)
    __builtin_amdgcn_s_wait_asynccnt(0);
#else
    asm volatile("s_wait_asynccnt 0x0" ::: "memory");
#endif
}

// Stage a 128-row x 32-elem bf16 panel (row stride ldg) into LDS (row stride 32).
// 512 x 16B chunks, 2 per thread for a 256-thread block.
DEV void stage_panel(const unsigned short* __restrict__ g, int ldg,
                     unsigned short* l, int tid) {
    #pragma unroll
    for (int i = 0; i < 2; ++i) {
        int c   = tid + i * 256;       // 0..511
        int row = c >> 2;
        int off = (c & 3) * 8;
        async_copy16(l + row * 32 + off, g + (size_t)row * ldg + off);
    }
}

// ---------- conversion kernels ----------
__global__ void k_f32_to_bf16(const float* __restrict__ in,
                              unsigned short* __restrict__ out, int n) {
    int i = blockIdx.x * blockDim.x + threadIdx.x;
    if (i < n) out[i] = f2bf(in[i]);
}

// W is [K][N] row-major f32; produce Wt [N][K] bf16 (column-contiguous K).
__global__ void k_transpose_to_bf16(const float* __restrict__ W,
                                    unsigned short* __restrict__ Wt,
                                    int K, int N) {
    int i = blockIdx.x * blockDim.x + threadIdx.x;
    if (i < K * N) {
        int k = i / N, n = i - k * N;
        Wt[(size_t)n * K + k] = f2bf(W[i]);
    }
}

// ---------- WMMA GEMM: C[M,N] = A[M,K](bf16) * Bt[N,K](bf16) + bias ----------
// Block = 256 threads = 8 waves (4x2), block tile 128x128, wave tile 32x64,
// K-step 32, double-buffered async global->LDS staging.
// mode 0: bf16 row-major [M][N]
// mode 1: bf16 [B,H,S,hd]   (Q,K projection;  m=b*2048+s, n=h*64+d)
// mode 2: bf16 [B,H,hd,S]   (V projection, transposed)
// mode 3: f32  row-major [M][N]  (final output)
__global__ void __launch_bounds__(256)
k_gemm_bf16(const unsigned short* __restrict__ A,
            const unsigned short* __restrict__ Bt,
            const float* __restrict__ bias,
            void* __restrict__ Out,
            int M, int N, int K, int mode) {
    __shared__ unsigned short As[2][128 * 32];   // 8 KB per buffer
    __shared__ unsigned short Bs[2][128 * 32];

    int tid  = threadIdx.x;
    int lane = tid & 31;
    int wv   = tid >> 5;           // 0..7
    int mw   = wv >> 1;            // 0..3  (32-row sub-tile)
    int nw   = wv & 1;             // 0..1  (64-col sub-tile)

    int nbx = N >> 7;              // blocks along N
    int bm  = blockIdx.x / nbx;
    int bn  = blockIdx.x - bm * nbx;
    int m0  = bm * 128, n0 = bn * 128;

    v8f acc[2][4];
    #pragma unroll
    for (int i = 0; i < 2; i++)
        #pragma unroll
        for (int t = 0; t < 4; t++) acc[i][t] = {};

    stage_panel(A  + (size_t)m0 * K, K, As[0], tid);
    stage_panel(Bt + (size_t)n0 * K, K, Bs[0], tid);
    wait_async0();
    __syncthreads();

    int NK = K >> 5;
    for (int kt = 0; kt < NK; ++kt) {
        int cur = kt & 1;
        if (kt + 1 < NK) {
            stage_panel(A  + (size_t)m0 * K + (kt + 1) * 32, K, As[cur ^ 1], tid);
            stage_panel(Bt + (size_t)n0 * K + (kt + 1) * 32, K, Bs[cur ^ 1], tid);
        }
        v16bf af[2];
        #pragma unroll
        for (int i = 0; i < 2; i++)
            af[i] = load_a_bf16(As[cur] + (mw * 32 + 16 * i) * 32, 32, lane);
        #pragma unroll
        for (int t = 0; t < 4; t++) {
            v16bf bf = load_b_bf16(Bs[cur] + (nw * 64 + 16 * t) * 32, 32, lane);
            acc[0][t] = wmma_bf16(af[0], bf, acc[0][t]);
            acc[1][t] = wmma_bf16(af[1], bf, acc[1][t]);
        }
        if (kt + 1 < NK) {
            wait_async0();
            asm volatile("s_wait_dscnt 0x0" ::: "memory");
            __syncthreads();
        }
    }

    int colLo = lane & 15, hiHalf = lane >> 4;
    #pragma unroll
    for (int i = 0; i < 2; i++) {
        #pragma unroll
        for (int t = 0; t < 4; t++) {
            int n = n0 + nw * 64 + 16 * t + colLo;
            float bv = bias[n];
            #pragma unroll
            for (int v = 0; v < 8; v++) {
                int m = m0 + mw * 32 + 16 * i + v + 8 * hiHalf;
                float val = acc[i][t][v] + bv;
                if (mode == 3) {
                    ((float*)Out)[(size_t)m * N + n] = val;
                } else {
                    unsigned short bfv = f2bf(val);
                    if (mode == 0) {
                        ((unsigned short*)Out)[(size_t)m * N + n] = bfv;
                    } else {
                        int b = m >> 11, s = m & 2047, h = n >> 6, d = n & 63;
                        if (mode == 1)
                            ((unsigned short*)Out)[(((size_t)(b * 16 + h) * 2048) + s) * 64 + d] = bfv;
                        else
                            ((unsigned short*)Out)[(((size_t)(b * 16 + h) * 64) + d) * 2048 + s] = bfv;
                    }
                }
            }
        }
    }
}

// ---------- causal flash attention ----------
// Q,K: [B,H,S,64] bf16 ; Vt: [B,H,64,S] bf16 ; Ctx: [B,S,1024] bf16
__global__ void __launch_bounds__(256)
k_flash_attn(const unsigned short* __restrict__ Q,
             const unsigned short* __restrict__ Kb,
             const unsigned short* __restrict__ Vt,
             unsigned short* __restrict__ Ctx) {
    const int S = 2048, HD = 64;
    __shared__ unsigned short ptile[8 * 16 * 32];   // per-wave 16x32 P tile

    int lane = threadIdx.x & 31;
    int wib  = threadIdx.x >> 5;
    unsigned short* lds = ptile + wib * (16 * 32);

    int wid = blockIdx.x * 8 + wib;        // 0 .. B*H*(S/16)-1 = 8191
    int qt  = wid & 127;                   // S/16 = 128 query tiles
    int bh  = wid >> 7;                    // b*16 + h
    int q0  = qt * 16;

    size_t base = (size_t)bh * S * HD;
    const unsigned short* Qp = Q  + base + (size_t)q0 * HD;
    const unsigned short* Kp = Kb + base;
    const unsigned short* Vp = Vt + base;  // [64][S]

    v16bf aq0 = load_a_bf16(Qp,      HD, lane);   // K-dim 0..31
    v16bf aq1 = load_a_bf16(Qp + 32, HD, lane);   // K-dim 32..63

    v8f o[4];
    #pragma unroll
    for (int t = 0; t < 4; t++) o[t] = {};
    v8f mrow, lrow;
    #pragma unroll
    for (int v = 0; v < 8; v++) { mrow[v] = -3.0e38f; lrow[v] = 0.f; }

    int hiHalf = lane >> 4, colLo = lane & 15;
    const float scale = 0.125f;            // 1/sqrt(64)
    int nkt = (q0 >> 5) + 1;               // causal: key tiles with c0 <= q0+15

    for (int kt = 0; kt < nkt; ++kt) {
        int c0 = kt * 32;
        // ---- scores: 16q x 32k, hd=64 via two chained WMMAs per half ----
        v8f s[2];
        #pragma unroll
        for (int n = 0; n < 2; n++) {
            v16bf blo = load_b_bf16(Kp + (size_t)(c0 + 16 * n) * HD,      HD, lane);
            v16bf bhi = load_b_bf16(Kp + (size_t)(c0 + 16 * n) * HD + 32, HD, lane);
            v8f z = {};
            z = wmma_bf16(aq0, blo, z);
            z = wmma_bf16(aq1, bhi, z);
            s[n] = z;
        }
        // ---- scale + causal mask (C layout: row = v+8*hiHalf, col = colLo) ----
        #pragma unroll
        for (int n = 0; n < 2; n++) {
            int col = c0 + 16 * n + colLo;
            #pragma unroll
            for (int v = 0; v < 8; v++) {
                int row = q0 + v + 8 * hiHalf;
                float x = s[n][v] * scale;
                s[n][v] = (col <= row) ? x : -1.0e30f;
            }
        }
        // ---- online softmax (row spans 16 lanes within each half-wave) ----
        #pragma unroll
        for (int v = 0; v < 8; v++) {
            float tm = fmaxf(s[0][v], s[1][v]);
            for (int msk = 1; msk < 16; msk <<= 1)
                tm = fmaxf(tm, __shfl_xor(tm, msk, 32));
            float mnew  = fmaxf(mrow[v], tm);
            float alpha = __expf(mrow[v] - mnew);
            float p0 = __expf(s[0][v] - mnew);
            float p1 = __expf(s[1][v] - mnew);
            float rs = p0 + p1;
            for (int msk = 1; msk < 16; msk <<= 1)
                rs += __shfl_xor(rs, msk, 32);
            lrow[v] = lrow[v] * alpha + rs;
            mrow[v] = mnew;
            #pragma unroll
            for (int t = 0; t < 4; t++) o[t][v] *= alpha;
            // P tile -> LDS (bf16, row-major 16x32) to re-swizzle C->A layout
            int row = v + 8 * hiHalf;
            lds[row * 32 + colLo]      = f2bf(p0);
            lds[row * 32 + 16 + colLo] = f2bf(p1);
        }
        asm volatile("s_wait_dscnt 0x0" ::: "memory");
        v16bf ap = load_a_bf16(lds, 32, lane);   // 16x32 A fragment of P
        // ---- PV: accumulate into 16x64 output tile ----
        #pragma unroll
        for (int t = 0; t < 4; t++) {
            v16bf bv = load_b_bf16(Vp + (size_t)(16 * t) * S + c0, S, lane);
            o[t] = wmma_bf16(ap, bv, o[t]);
        }
    }

    // ---- normalize and store context as [B,S,H*hd] bf16 ----
    int b = bh >> 4, h = bh & 15;
    #pragma unroll
    for (int v = 0; v < 8; v++) {
        float inv = 1.0f / lrow[v];
        int qrow = q0 + v + 8 * hiHalf;
        size_t rowbase = ((size_t)b * 2048 + qrow) * 1024 + h * 64;
        #pragma unroll
        for (int t = 0; t < 4; t++)
            Ctx[rowbase + 16 * t + colLo] = f2bf(o[t][v] * inv);
    }
}

// ---------- host launch ----------
extern "C" void kernel_launch(void* const* d_in, const int* in_sizes, int n_in,
                              void* d_out, int out_size, void* d_ws, size_t ws_size,
                              hipStream_t stream) {
    const float* x  = (const float*)d_in[0];
    const float* Wq = (const float*)d_in[1];
    const float* bq = (const float*)d_in[2];
    const float* Wk = (const float*)d_in[3];
    const float* bk = (const float*)d_in[4];
    const float* Wv = (const float*)d_in[5];
    const float* bv = (const float*)d_in[6];
    const float* Wo = (const float*)d_in[7];
    const float* bo = (const float*)d_in[8];
    float* out = (float*)d_out;

    const int S = 2048, D = 1024;
    const int M = 4 * S;                       // 8192 rows

    char* w = (char*)d_ws;
    unsigned short* xb  = (unsigned short*)w;  w += (size_t)M * D * 2;   // 16 MB
    unsigned short* Wqt = (unsigned short*)w;  w += (size_t)D * D * 2;
    unsigned short* Wkt = (unsigned short*)w;  w += (size_t)D * D * 2;
    unsigned short* Wvt = (unsigned short*)w;  w += (size_t)D * D * 2;
    unsigned short* Wot = (unsigned short*)w;  w += (size_t)D * D * 2;
    unsigned short* Qb  = (unsigned short*)w;  w += (size_t)M * D * 2;   // [B,H,S,64]
    unsigned short* Kbf = (unsigned short*)w;  w += (size_t)M * D * 2;   // [B,H,S,64]
    unsigned short* Vtb = (unsigned short*)w;  w += (size_t)M * D * 2;   // [B,H,64,S]
    unsigned short* ctx = (unsigned short*)w;  w += (size_t)M * D * 2;   // [B,S,D]

    // 1) cast activations and transpose+cast weights to bf16
    k_f32_to_bf16<<<(M * D) / 256, 256, 0, stream>>>(x, xb, M * D);
    k_transpose_to_bf16<<<(D * D) / 256, 256, 0, stream>>>(Wq, Wqt, D, D);
    k_transpose_to_bf16<<<(D * D) / 256, 256, 0, stream>>>(Wk, Wkt, D, D);
    k_transpose_to_bf16<<<(D * D) / 256, 256, 0, stream>>>(Wv, Wvt, D, D);
    k_transpose_to_bf16<<<(D * D) / 256, 256, 0, stream>>>(Wo, Wot, D, D);

    // 2) Q/K/V projections: (M/128)*(D/128) = 64*8 = 512 blocks
    const int gemmBlocks = (M / 128) * (D / 128);
    k_gemm_bf16<<<gemmBlocks, 256, 0, stream>>>(xb, Wqt, bq, Qb,  M, D, D, 1);
    k_gemm_bf16<<<gemmBlocks, 256, 0, stream>>>(xb, Wkt, bk, Kbf, M, D, D, 1);
    k_gemm_bf16<<<gemmBlocks, 256, 0, stream>>>(xb, Wvt, bv, Vtb, M, D, D, 2);

    // 3) causal flash attention; waves = B*H*(S/16) = 8192, 8 waves/block
    k_flash_attn<<<1024, 256, 0, stream>>>(Qb, Kbf, Vtb, ctx);

    // 4) output projection -> f32
    k_gemm_bf16<<<gemmBlocks, 256, 0, stream>>>(ctx, Wot, bo, (void*)out, M, D, D, 3);
}
